// MessageGraphAttention_3418793968215
// MI455X (gfx1250) — matrix operand order repaired and checked by reading, not compiled, not run on hardware
//
#include <hip/hip_runtime.h>
#include <hip/hip_fp16.h>
#include <hip/hip_bf16.h>

// MessageGraphAttention, MI455X (gfx1250, wave32).
//   out[a] = sum_b relu(A[a,b]*(X1[a]+X2[b]+E[a,b]@W3)+bias)
//                 * sigmoid(A[a,b]*(Y1[a]+Y2[b]+E[a,b]@Wa3)+bias)
// Heavy part (E@W3, E@Wa3) runs on v_wmma_f32_16x16x32_f16 (f16 in, f32 acc).
// HBM-bound: 134 MB of E streamed once, coalesced -> ~6us floor at 23.3 TB/s.

typedef __attribute__((ext_vector_type(16))) _Float16 v16h;
typedef __attribute__((ext_vector_type(8)))  float    v8f;

#define NN      1024
#define FNODE   64
#define FEDGE   32

// ---------------------------------------------------------------------------
// Kernel 1: tiny precompute of X1=H@W[0:64], X2=H@W[64:128], Y1/Y2 with W_att.
// 65536 threads, each one (node,feature) pair; 33 MFLOP total.
// ws layout (floats): X1 @0, X2 @65536, Y1 @131072, Y2 @196608  (1 MB).
// ---------------------------------------------------------------------------
__global__ void mga_precompute(const float* __restrict__ H,
                               const float* __restrict__ W,
                               const float* __restrict__ Wa,
                               float* __restrict__ ws) {
    int gid = blockIdx.x * blockDim.x + threadIdx.x;   // 0 .. 1024*64-1
    int i = gid >> 6;    // node index
    int j = gid & 63;    // output feature
    const float* h = H + i * FNODE;
    float x1 = 0.f, x2 = 0.f, y1 = 0.f, y2 = 0.f;
#pragma unroll 8
    for (int k = 0; k < FNODE; ++k) {
        float hv = h[k];
        x1 += hv * W [ k        * FNODE + j];
        x2 += hv * W [(FNODE+k) * FNODE + j];
        y1 += hv * Wa[ k        * FNODE + j];
        y2 += hv * Wa[(FNODE+k) * FNODE + j];
    }
    ws[            gid] = x1;
    ws[ 65536 +    gid] = x2;
    ws[131072 +    gid] = y1;
    ws[196608 +    gid] = y2;
}

// ---------------------------------------------------------------------------
// Kernel 2: one block per row `a`, 8 waves; each wave handles 8 b-tiles of 16.
// ---------------------------------------------------------------------------
__global__ __launch_bounds__(256)
void mga_main(const float* __restrict__ A,
              const float* __restrict__ E,
              const float* __restrict__ W,
              const float* __restrict__ Wa,
              const float* __restrict__ bias,
              const float* __restrict__ ws,
              float* __restrict__ out) {
    const int a     = blockIdx.x;
    const int tid   = threadIdx.x;
    const int lane  = tid & 31;
    const int wave  = tid >> 5;
    const int nlo   = lane & 15;     // column (B/C/D) or row (A) within tile
    const int khalf = lane >> 4;     // which K-half this lane carries

    const float* X1 = ws;
    const float* X2 = ws +  65536;
    const float* Y1 = ws + 131072;
    const float* Y2 = ws + 196608;

    // ---- B fragments: W3 = W[128:160,:], Wa3 = Wa[128:160,:], 4 column tiles
    // ISA B layout (32x16, f16): column n = lane%16, K = 8*(lane/16) + h (h<8), 16 + 8*(lane/16) + (h-8) (h>=8).
    v16h bw[4], ba[4];
#pragma unroll
    for (int ct = 0; ct < 4; ++ct) {
        const int n = ct * 16 + nlo;
#pragma unroll
        for (int h = 0; h < 16; ++h) {
            const int k = khalf * 8 + h + (h >= 8 ? 8 : 0);
            bw[ct][h] = (_Float16)W [(2 * FNODE + k) * FNODE + n];
            ba[ct][h] = (_Float16)Wa[(2 * FNODE + k) * FNODE + n];
        }
    }

    // Loop-invariant per-lane scalars for column j = ct*16 + nlo.
    float x1v[4], y1v[4], bv[4];
#pragma unroll
    for (int ct = 0; ct < 4; ++ct) {
        const int j = ct * 16 + nlo;
        x1v[ct] = X1[a * FNODE + j];
        y1v[ct] = Y1[a * FNODE + j];
        bv[ct]  = bias[j];
    }

    // C/D layout: VGPR r holds M=r (lanes 0-15) / M=r+8 (lanes 16-31).
    const int mbase = (lane < 16) ? 0 : 8;

    float acc[4] = {0.f, 0.f, 0.f, 0.f};

    for (int t = wave; t < NN / 16; t += 8) {
        const int b0 = t * 16;

        // ---- A fragment: 16 rows of E (rows = b index, M dim), K = 32.
        // ISA A layout (16x32, f16): row m = lane%16; halves 0-7 hold
        // K = 8*khalf .. 8*khalf+7, halves 8-15 hold K = 16+8*khalf .. +7.
        const size_t row = (size_t)a * NN + (size_t)(b0 + nlo);
        const float* ep  = E + row * FEDGE + khalf * 8;
        const float4* p  = reinterpret_cast<const float4*>(ep);
        const float4 q0 = p[0], q1 = p[1];      // K chunk [8k .. 8k+7]
        const float4 q2 = p[4], q3 = p[5];      // K chunk [16+8k .. 16+8k+7]
        v16h af;
        af[0]  = (_Float16)q0.x; af[1]  = (_Float16)q0.y;
        af[2]  = (_Float16)q0.z; af[3]  = (_Float16)q0.w;
        af[4]  = (_Float16)q1.x; af[5]  = (_Float16)q1.y;
        af[6]  = (_Float16)q1.z; af[7]  = (_Float16)q1.w;
        af[8]  = (_Float16)q2.x; af[9]  = (_Float16)q2.y;
        af[10] = (_Float16)q2.z; af[11] = (_Float16)q2.w;
        af[12] = (_Float16)q3.x; af[13] = (_Float16)q3.y;
        af[14] = (_Float16)q3.z; af[15] = (_Float16)q3.w;

        // Adjacency values for the 8 rows this lane's accumulators see.
        float avals[8];
#pragma unroll
        for (int r = 0; r < 8; ++r)
            avals[r] = A[(size_t)a * NN + (size_t)(b0 + mbase + r)];

#pragma unroll
        for (int ct = 0; ct < 4; ++ct) {
            v8f cz = {};
            v8f d  = __builtin_amdgcn_wmma_f32_16x16x32_f16(
                         false, af, false, bw[ct], (short)0, cz, false, false);
            v8f da = __builtin_amdgcn_wmma_f32_16x16x32_f16(
                         false, af, false, ba[ct], (short)0, cz, false, false);
            const int j = ct * 16 + nlo;
            float s = 0.f;
#pragma unroll
            for (int r = 0; r < 8; ++r) {
                const int b   = b0 + mbase + r;
                const float g = avals[r];
                float hp = g * (x1v[ct] + X2[b * FNODE + j] + d[r])  + bv[ct];
                float ha = g * (y1v[ct] + Y2[b * FNODE + j] + da[r]) + bv[ct];
                s += fmaxf(hp, 0.f) * (1.f / (1.f + __expf(-ha)));
            }
            acc[ct] += s;
        }
    }

    // ---- reduction: lane l pairs with l^16 (same column j), then across waves
    __shared__ float red[8][FNODE];
#pragma unroll
    for (int ct = 0; ct < 4; ++ct) {
        float o = acc[ct] + __shfl_xor(acc[ct], 16, 32);
        if (lane < 16) red[wave][ct * 16 + lane] = o;
    }
    __syncthreads();
    if (tid < FNODE) {
        float s = 0.f;
#pragma unroll
        for (int w = 0; w < 8; ++w) s += red[w][tid];
        out[a * FNODE + tid] = s;   // each `a` owned by exactly one block
    }
}

// ---------------------------------------------------------------------------
extern "C" void kernel_launch(void* const* d_in, const int* in_sizes, int n_in,
                              void* d_out, int out_size, void* d_ws, size_t ws_size,
                              hipStream_t stream) {
    (void)in_sizes; (void)n_in; (void)out_size; (void)ws_size;
    const float* H    = (const float*)d_in[0];   // (1024, 64)
    const float* A    = (const float*)d_in[1];   // (1024, 1024)
    const float* E    = (const float*)d_in[2];   // (1024, 1024, 32)
    const float* W    = (const float*)d_in[3];   // (160, 64)
    const float* Wa   = (const float*)d_in[4];   // (160, 64)
    const float* bias = (const float*)d_in[5];   // (64,)
    float*       ws   = (float*)d_ws;            // needs 1 MB
    float*       out  = (float*)d_out;           // (1024, 64)

    mga_precompute<<<(NN * FNODE) / 256, 256, 0, stream>>>(H, W, Wa, ws);
    mga_main<<<NN, 256, 0, stream>>>(A, E, W, Wa, bias, ws, out);
}
